// LightAlignedMemoryBank_82197084111492
// MI455X (gfx1250) — matrix operand order repaired and checked by reading, compiled, not run on hardware
//
#include <hip/hip_runtime.h>
#include <hip/hip_bf16.h>
#include <math.h>

typedef __attribute__((ext_vector_type(16))) _Float16 v16h;
typedef __attribute__((ext_vector_type(8)))  float    v8f;

#define CCH 256
#define HH 100
#define WW 100
#define NP 10000          // H*W
#define NT 25             // deform taps
#define MT33 16           // 256/16 output-channel tiles
#define KSTEPS33 72       // 8 ci-blocks * 9 taps
#define EPSV 1e-5f

// CDNA5 async global->LDS copy (16B per active lane), tracked by ASYNCcnt.
// VDST = per-lane LDS byte address, VADDR = per-lane 64-bit global address.
__device__ __forceinline__ void async_ld16(unsigned ldsoff, const _Float16* g) {
  asm volatile("global_load_async_to_lds_b128 %0, %1, off"
               :: "v"(ldsoff), "v"(g) : "memory");
}

// ---------------------------------------------------------------------------
// Transpose [256][10000] f32  ->  [10000][256] f16   (LDS tiled, coalesced)
// grid: (157, 8), block 256
__global__ void k_transpose_cf16(const float* __restrict__ in, _Float16* __restrict__ out) {
  __shared__ float tile[32][65];
  int c0 = blockIdx.y * 32;
  int p0 = blockIdx.x * 64;
  int tid = threadIdx.x;
  for (int t = 0; t < 8; ++t) {
    int idx = t * 256 + tid;
    int ci = idx >> 6, pj = idx & 63;
    int p = p0 + pj;
    tile[ci][pj] = (p < NP) ? in[(c0 + ci) * NP + p] : 0.f;
  }
  __syncthreads();
  for (int t = 0; t < 8; ++t) {
    int idx = t * 256 + tid;
    int pj = idx >> 5, ci = idx & 31;
    int p = p0 + pj;
    if (p < NP) out[p * CCH + c0 + ci] = (_Float16)tile[ci][pj];
  }
}

// ---------------------------------------------------------------------------
// Pack conv3x3 weights [256][256][3][3] f32 into WMMA A-tiles:
// pA[mt][kt=cb*9+rs][lane][h] f16, k = 16*(lane/16) + h within each 32-ci block (contiguous order).
// grid: 2304 blocks * 256
__global__ void k_pack_w33(const float* __restrict__ w, _Float16* __restrict__ pA) {
  int idx = blockIdx.x * 256 + threadIdx.x;            // ((mt*72+kt)*32+lane)*16+h
  int h    = idx & 15;
  int lane = (idx >> 4) & 31;
  int kt   = (idx >> 9) % KSTEPS33;
  int mt   = (idx >> 9) / KSTEPS33;
  int cb = kt / 9, rs = kt % 9;
  int k32 = h + ((lane >= 16) ? 16 : 0);
  int m  = mt * 16 + (lane & 15);
  int ci = cb * 32 + k32;
  pA[idx] = (_Float16)w[(m * CCH + ci) * 9 + rs];
}

// Pack offset conv1x1 weights [50][512] -> pA[4][16][32][16] (rows >= 50 zero)
// grid: 128 blocks * 256
__global__ void k_pack_woff(const float* __restrict__ w, _Float16* __restrict__ pA) {
  int idx = blockIdx.x * 256 + threadIdx.x;
  int h    = idx & 15;
  int lane = (idx >> 4) & 31;
  int kt   = (idx >> 9) & 15;
  int mt   = (idx >> 9) >> 4;
  int k32 = h + ((lane >= 16) ? 16 : 0);
  int m = mt * 16 + (lane & 15);
  int k = kt * 32 + k32;
  pA[idx] = (m < 50) ? (_Float16)w[m * 512 + k] : (_Float16)0.f;
}

// ---------------------------------------------------------------------------
// Offset conv1x1 via WMMA: out[m][p] = sum_k A[m][k] * X[k][p] + bias[m]
// K=512 (cur 0..255, hist 256..511). grid 625 blocks, 128 threads (4 waves).
__global__ void __launch_bounds__(128) k_conv1x1_wmma(
    const _Float16* __restrict__ curT, const _Float16* __restrict__ histT,
    const _Float16* __restrict__ pA, const float* __restrict__ bias,
    float* __restrict__ out) {
  __shared__ __attribute__((aligned(64))) _Float16 xs[16][32];   // [px][ci]
  int tid = threadIdx.x, wave = tid >> 5, lane = tid & 31;
  int p0 = blockIdx.x * 16;
  v8f acc = {};
  for (int cb = 0; cb < 16; ++cb) {
    __syncthreads();
    if (tid < 32) {
      int px = tid >> 1, part = tid & 1;
      const _Float16* src = (cb < 8) ? curT : histT;
      int cc = (cb & 7) * 32;
      *(v16h*)(&xs[px][part * 16]) =
          *(const v16h*)(src + (p0 + px) * CCH + cc + part * 16);
    }
    __syncthreads();
    int n = lane & 15, g = lane >> 4;
    v16h b = *(const v16h*)(&xs[n][g * 16]);
    v16h a = *(const v16h*)(pA + (size_t)((wave * 16 + cb) * 32 + lane) * 16);
    acc = __builtin_amdgcn_wmma_f32_16x16x32_f16(false, a, false, b, (short)0, acc, false, false);
  }
  int n = lane & 15;
  int p = p0 + n;
  int mbase = wave * 16 + ((lane >= 16) ? 8 : 0);
  for (int r = 0; r < 8; ++r) {
    int m = mbase + r;
    if (m < 50) out[m * NP + p] = acc[r] + bias[m];
  }
}

// ---------------------------------------------------------------------------
// Depthwise deformable conv sampling. One block per pixel, thread = channel.
// histT is [p][c] f16 (channel-contiguous -> coalesced gathers).
__global__ void __launch_bounds__(256) k_deform(
    const _Float16* __restrict__ histT, const float* __restrict__ off,
    const float* __restrict__ dw, float* __restrict__ aligned) {
  int p = blockIdx.x, c = threadIdx.x;
  int py = p / WW, px = p % WW;
  __shared__ int   si00[NT], si01[NT], si10[NT], si11[NT];
  __shared__ float sw00[NT], sw01[NT], sw10[NT], sw11[NT];
  if (c < NT) {
    float dy = off[(2 * c + 0) * NP + p];
    float dx = off[(2 * c + 1) * NP + p];
    float yy = (float)py + (float)(c / 5 - 2) + dy;
    float xx = (float)px + (float)(c % 5 - 2) + dx;
    float y0f = floorf(yy), x0f = floorf(xx);
    int y0 = (int)y0f, x0 = (int)x0f;
    float wy = yy - y0f, wx = xx - x0f;
    int y1 = y0 + 1, x1 = x0 + 1;
    float v00 = (y0 >= 0 && y0 < HH && x0 >= 0 && x0 < WW) ? 1.f : 0.f;
    float v01 = (y0 >= 0 && y0 < HH && x1 >= 0 && x1 < WW) ? 1.f : 0.f;
    float v10 = (y1 >= 0 && y1 < HH && x0 >= 0 && x0 < WW) ? 1.f : 0.f;
    float v11 = (y1 >= 0 && y1 < HH && x1 >= 0 && x1 < WW) ? 1.f : 0.f;
    int yi0 = min(max(y0, 0), HH - 1), yi1 = min(max(y1, 0), HH - 1);
    int xi0 = min(max(x0, 0), WW - 1), xi1 = min(max(x1, 0), WW - 1);
    si00[c] = yi0 * WW + xi0;  sw00[c] = (1.f - wy) * (1.f - wx) * v00;
    si01[c] = yi0 * WW + xi1;  sw01[c] = (1.f - wy) * wx * v01;
    si10[c] = yi1 * WW + xi0;  sw10[c] = wy * (1.f - wx) * v10;
    si11[c] = yi1 * WW + xi1;  sw11[c] = wy * wx * v11;
  }
  __syncthreads();
  const _Float16* hc = histT + c;
  float acc = 0.f;
  for (int t = 0; t < NT; ++t) {
    float v = sw00[t] * (float)hc[(size_t)si00[t] * CCH]
            + sw01[t] * (float)hc[(size_t)si01[t] * CCH]
            + sw10[t] * (float)hc[(size_t)si10[t] * CCH]
            + sw11[t] * (float)hc[(size_t)si11[t] * CCH];
    acc += dw[c * NT + t] * v;
  }
  aligned[c * NP + p] = acc;
}

// ---------------------------------------------------------------------------
// Per-channel reductions: pooled = mean(cur - aligned), diffm = mean|cur-hist|
// grid 256 blocks (one per channel)
__global__ void k_reduce_pd(const float* __restrict__ cur, const float* __restrict__ aligned,
                            const float* __restrict__ hist,
                            float* __restrict__ pooled, float* __restrict__ diffm) {
  int c = blockIdx.x, tid = threadIdx.x;
  float s1 = 0.f, s2 = 0.f;
  for (int p = tid; p < NP; p += 256) {
    float cv = cur[c * NP + p];
    s1 += cv - aligned[c * NP + p];
    s2 += fabsf(cv - hist[c * NP + p]);
  }
  __shared__ float r1[256], r2[256];
  r1[tid] = s1; r2[tid] = s2;
  __syncthreads();
  for (int st = 128; st > 0; st >>= 1) {
    if (tid < st) { r1[tid] += r1[tid + st]; r2[tid] += r2[tid + st]; }
    __syncthreads();
  }
  if (tid == 0) { pooled[c] = r1[0] * (1.f / NP); diffm[c] = r2[0] * (1.f / NP); }
}

// Channel attention (256->32->256) + frame weight. 1 block, 256 threads.
__global__ void k_attn(const float* __restrict__ pooled, const float* __restrict__ diffm,
                       const float* __restrict__ w1, const float* __restrict__ w2,
                       float* __restrict__ attn, float* __restrict__ wgt,
                       float* __restrict__ wsum, int init) {
  __shared__ float sp[256], sh[32];
  int tid = threadIdx.x;
  sp[tid] = pooled[tid];
  __syncthreads();
  if (tid < 32) {
    float a = 0.f;
    for (int c2 = 0; c2 < 256; ++c2) a += w1[tid * 256 + c2] * sp[c2];
    sh[tid] = fmaxf(a, 0.f);
  }
  __syncthreads();
  float a = 0.f;
  for (int m = 0; m < 32; ++m) a += w2[tid * 32 + m] * sh[m];
  attn[tid] = 1.f / (1.f + expf(-a));
  float wv = 1.f / (1.f + expf(diffm[tid]));   // sigmoid(-diff)
  wgt[tid] = wv;
  wsum[tid] = init ? wv : (wsum[tid] + wv);
}

// fusedT[p][c] = f16(cur + aligned*attn).  grid 256 blocks (c)
__global__ void k_build_fused(const float* __restrict__ cur, const float* __restrict__ aligned,
                              const float* __restrict__ attn, _Float16* __restrict__ fT) {
  int c = blockIdx.x;
  float a = attn[c];
  for (int p = threadIdx.x; p < NP; p += 256)
    fT[p * CCH + c] = (_Float16)(cur[c * NP + p] + aligned[c * NP + p] * a);
}

// fusedT[p][c] = f16(accum / (wsum + 1e-6))
__global__ void k_build_final(const float* __restrict__ accum, const float* __restrict__ wsum,
                              _Float16* __restrict__ fT) {
  int c = blockIdx.x;
  float inv = 1.f / (wsum[c] + 1e-6f);
  for (int p = threadIdx.x; p < NP; p += 256)
    fT[p * CCH + c] = (_Float16)(accum[c * NP + p] * inv);
}

// ---------------------------------------------------------------------------
// conv3x3 (256->256, SAME, zero pad) as implicit-GEMM WMMA with
// double-buffered LDS halo tiles filled by async global->LDS copies.
// Workgroup: 1 row x 16 pixels x all 256 Co. 8 waves x 2 M-tiles.
// grid 700 (= 100 rows * 7 x-tiles), block 256.
__global__ void __launch_bounds__(256) k_conv3x3_wmma(
    const _Float16* __restrict__ xT,   // [10000][256]
    const _Float16* __restrict__ pA,   // [16][72][32][16]
    float* __restrict__ out) {         // [256][10000]
  __shared__ __attribute__((aligned(64))) _Float16 xs[2][3][18][32];  // buf, row, col, ci
  int tid = threadIdx.x, wave = tid >> 5, lane = tid & 31;
  int ty = blockIdx.x / 7;
  int x0 = (blockIdx.x % 7) * 16;
  int mt0 = wave * 2;

  // staging lanes: 54 (row,col) halo slots * 2 16-half chunks = 108 tasks
  int slot = tid >> 1, part = tid & 1;
  int row = slot / 18, col = slot % 18;
  int yy = ty + row - 1, xx = x0 + col - 1;
  bool issuer = (tid < 108);
  bool inb = issuer && (yy >= 0) && (yy < HH) && (xx >= 0) && (xx < WW);
  int pidx = inb ? (yy * WW + xx) : 0;
  const _Float16* gbase = xT + (size_t)pidx * CCH + part * 16;
  unsigned lds0 = (unsigned)(size_t)(&xs[0][row][col][part * 16]);
  unsigned lds1 = (unsigned)(size_t)(&xs[1][row][col][part * 16]);

  // zero-pad boundary slots once (cb-invariant), in both buffers
  if (issuer && !inb) {
    v16h z = {};
    *(v16h*)(&xs[0][row][col][part * 16]) = z;
    *(v16h*)(&xs[1][row][col][part * 16]) = z;
  }
  // prefetch ci-block 0 into buffer 0
  if (inb) {
    async_ld16(lds0,      gbase);
    async_ld16(lds0 + 16, gbase + 8);
  }

  v8f acc0 = {}, acc1 = {};
  int n = lane & 15, hgrp = lane >> 4;
#pragma unroll
  for (int cb = 0; cb < 8; ++cb) {
    if (cb < 7) {
      if (inb) {                                   // prefetch next ci-block
        const _Float16* gp = gbase + (cb + 1) * 32;
        unsigned ldsn = ((cb + 1) & 1) ? lds1 : lds0;
        async_ld16(ldsn,      gp);
        async_ld16(ldsn + 16, gp + 8);
      }
      asm volatile("s_wait_asynccnt 0x2" ::: "memory");  // oldest pair (cb) done
    } else {
      asm volatile("s_wait_asynccnt 0x0" ::: "memory");
    }
    __syncthreads();
    const _Float16* xb = &xs[cb & 1][0][0][0];
#pragma unroll
    for (int rs = 0; rs < 9; ++rs) {
      int r = rs / 3, s = rs % 3;
      v16h b = *(const v16h*)(xb + (size_t)((r * 18 + s + n) * 32 + hgrp * 16));
      int kt = cb * 9 + rs;
      v16h a0 = *(const v16h*)(pA + (size_t)((mt0 * KSTEPS33 + kt) * 32 + lane) * 16);
      v16h a1 = *(const v16h*)(pA + (size_t)(((mt0 + 1) * KSTEPS33 + kt) * 32 + lane) * 16);
      acc0 = __builtin_amdgcn_wmma_f32_16x16x32_f16(false, a0, false, b, (short)0, acc0, false, false);
      acc1 = __builtin_amdgcn_wmma_f32_16x16x32_f16(false, a1, false, b, (short)0, acc1, false, false);
    }
    __syncthreads();
  }
  int xo = x0 + n;
  if (xo < WW) {
    int p = ty * WW + xo;
    int mbase = (lane >= 16) ? 8 : 0;
    for (int r = 0; r < 8; ++r) {
      out[(size_t)(mt0 * 16 + mbase + r) * NP + p]       = acc0[r];
      out[(size_t)((mt0 + 1) * 16 + mbase + r) * NP + p] = acc1[r];
    }
  }
}

// ---------------------------------------------------------------------------
// GroupNorm stats (8 groups of 32 channels).  grid 8 blocks.
__global__ void k_gn_stats(const float* __restrict__ y, float* __restrict__ st) {
  int g = blockIdx.x, tid = threadIdx.x;
  float s = 0.f, s2 = 0.f;
  const float* base = y + (size_t)g * 32 * NP;
  for (int i = tid; i < 32 * NP; i += 256) { float v = base[i]; s += v; s2 += v * v; }
  __shared__ float r1[256], r2[256];
  r1[tid] = s; r2[tid] = s2;
  __syncthreads();
  for (int stp = 128; stp > 0; stp >>= 1) {
    if (tid < stp) { r1[tid] += r1[tid + stp]; r2[tid] += r2[tid + stp]; }
    __syncthreads();
  }
  if (tid == 0) {
    float m = r1[0] / (32.f * NP);
    st[g * 2] = m;
    st[g * 2 + 1] = r2[0] / (32.f * NP) - m * m;
  }
}

// GN apply + relu + weighted accumulate.  grid 256 blocks (c).
__global__ void k_gn_apply_acc(const float* __restrict__ y, const float* __restrict__ st,
                               const float* __restrict__ gamma, const float* __restrict__ beta,
                               const float* __restrict__ wgt, float* __restrict__ accum, int init) {
  int c = blockIdx.x, g = c >> 5;
  float m = st[g * 2], v = st[g * 2 + 1];
  float inv = rsqrtf(v + EPSV);
  float ga = gamma[c], be = beta[c], w = wgt[c];
  for (int p = threadIdx.x; p < NP; p += 256) {
    float val = (y[c * NP + p] - m) * inv * ga + be;
    val = fmaxf(val, 0.f) * w;
    accum[c * NP + p] = init ? val : accum[c * NP + p] + val;
  }
}

// BatchNorm stats per channel (B=1 -> over spatial).  grid 256 blocks.
__global__ void k_bn_stats(const float* __restrict__ y, float* __restrict__ st) {
  int c = blockIdx.x, tid = threadIdx.x;
  float s = 0.f, s2 = 0.f;
  for (int p = tid; p < NP; p += 256) { float v = y[c * NP + p]; s += v; s2 += v * v; }
  __shared__ float r1[256], r2[256];
  r1[tid] = s; r2[tid] = s2;
  __syncthreads();
  for (int stp = 128; stp > 0; stp >>= 1) {
    if (tid < stp) { r1[tid] += r1[tid + stp]; r2[tid] += r2[tid + stp]; }
    __syncthreads();
  }
  if (tid == 0) {
    float m = r1[0] / (float)NP;
    st[c * 2] = m;
    st[c * 2 + 1] = r2[0] / (float)NP - m * m;
  }
}

// BN apply + relu -> d_out.  grid 256 blocks.
__global__ void k_bn_apply(const float* __restrict__ y, const float* __restrict__ st,
                           const float* __restrict__ g, const float* __restrict__ b,
                           float* __restrict__ out) {
  int c = blockIdx.x;
  float m = st[c * 2], v = st[c * 2 + 1];
  float inv = rsqrtf(v + EPSV);
  float ga = g[c], be = b[c];
  for (int p = threadIdx.x; p < NP; p += 256) {
    float val = (y[c * NP + p] - m) * inv * ga + be;
    out[c * NP + p] = fmaxf(val, 0.f);
  }
}

// ---------------------------------------------------------------------------
extern "C" void kernel_launch(void* const* d_in, const int* in_sizes, int n_in,
                              void* d_out, int out_size, void* d_ws, size_t ws_size,
                              hipStream_t stream) {
  const float* cur      = (const float*)d_in[0];
  const float* hist0    = (const float*)d_in[1];
  const float* hist1    = (const float*)d_in[2];
  const float* offset_w = (const float*)d_in[3];
  const float* offset_b = (const float*)d_in[4];
  const float* deform_w = (const float*)d_in[5];
  const float* ca_w1    = (const float*)d_in[6];
  const float* ca_w2    = (const float*)d_in[7];
  const float* out_w    = (const float*)d_in[8];
  const float* gn_g     = (const float*)d_in[9];
  const float* gn_b     = (const float*)d_in[10];
  const float* fuse_w   = (const float*)d_in[11];
  const float* bn_g     = (const float*)d_in[12];
  const float* bn_b     = (const float*)d_in[13];
  float* outp = (float*)d_out;

  char* wsp = (char*)d_ws;
  auto alloc = [&](size_t bytes) -> void* {
    void* p = (void*)wsp;
    wsp += (bytes + 255) & ~(size_t)255;
    return p;
  };
  _Float16* curT    = (_Float16*)alloc((size_t)NP * CCH * 2);
  _Float16* h0T     = (_Float16*)alloc((size_t)NP * CCH * 2);
  _Float16* h1T     = (_Float16*)alloc((size_t)NP * CCH * 2);
  _Float16* fusedT  = (_Float16*)alloc((size_t)NP * CCH * 2);
  _Float16* pA_ha   = (_Float16*)alloc((size_t)MT33 * KSTEPS33 * 32 * 16 * 2);
  _Float16* pA_fu   = (_Float16*)alloc((size_t)MT33 * KSTEPS33 * 32 * 16 * 2);
  _Float16* pA_off  = (_Float16*)alloc((size_t)4 * 16 * 32 * 16 * 2);
  float* off_buf = (float*)alloc((size_t)50 * NP * 4);
  float* aligned = (float*)alloc((size_t)CCH * NP * 4);
  float* convout = (float*)alloc((size_t)CCH * NP * 4);
  float* accum   = (float*)alloc((size_t)CCH * NP * 4);
  float* pooled  = (float*)alloc(CCH * 4);
  float* diffm   = (float*)alloc(CCH * 4);
  float* attn    = (float*)alloc(CCH * 4);
  float* wgt     = (float*)alloc(CCH * 4);
  float* wsum    = (float*)alloc(CCH * 4);
  float* gnstats = (float*)alloc(8 * 2 * 4);
  float* bnstats = (float*)alloc(CCH * 2 * 4);

  dim3 tgrid(157, 8);
  k_transpose_cf16<<<tgrid, 256, 0, stream>>>(cur,   curT);
  k_transpose_cf16<<<tgrid, 256, 0, stream>>>(hist0, h0T);
  k_transpose_cf16<<<tgrid, 256, 0, stream>>>(hist1, h1T);
  k_pack_w33<<<2304, 256, 0, stream>>>(out_w,  pA_ha);
  k_pack_w33<<<2304, 256, 0, stream>>>(fuse_w, pA_fu);
  k_pack_woff<<<128, 256, 0, stream>>>(offset_w, pA_off);

  for (int hidx = 0; hidx < 2; ++hidx) {
    const float*    hist  = (hidx == 0) ? hist0 : hist1;
    const _Float16* histT = (hidx == 0) ? h0T : h1T;
    int init = (hidx == 0) ? 1 : 0;

    k_conv1x1_wmma<<<625, 128, 0, stream>>>(curT, histT, pA_off, offset_b, off_buf);
    k_deform<<<NP, 256, 0, stream>>>(histT, off_buf, deform_w, aligned);
    k_reduce_pd<<<CCH, 256, 0, stream>>>(cur, aligned, hist, pooled, diffm);
    k_attn<<<1, 256, 0, stream>>>(pooled, diffm, ca_w1, ca_w2, attn, wgt, wsum, init);
    k_build_fused<<<CCH, 256, 0, stream>>>(cur, aligned, attn, fusedT);
    k_conv3x3_wmma<<<700, 256, 0, stream>>>(fusedT, pA_ha, convout);
    k_gn_stats<<<8, 256, 0, stream>>>(convout, gnstats);
    k_gn_apply_acc<<<CCH, 256, 0, stream>>>(convout, gnstats, gn_g, gn_b, wgt, accum, init);
  }

  k_build_final<<<CCH, 256, 0, stream>>>(accum, wsum, fusedT);
  k_conv3x3_wmma<<<700, 256, 0, stream>>>(fusedT, pA_fu, convout);
  k_bn_stats<<<CCH, 256, 0, stream>>>(convout, bnstats);
  k_bn_apply<<<CCH, 256, 0, stream>>>(convout, bnstats, bn_g, bn_b, outp);
}
